// sLSTM_57312043597976
// MI455X (gfx1250) — matrix operand order repaired and checked by reading, not compiled, not run on hardware
//
#include <hip/hip_runtime.h>
#include <hip/hip_bf16.h>

// ---------------- problem constants ----------------
#define BB       8192
#define DIN      2048
#define DD       2048
#define HH       8
#define HDIM     256
#define PROJ     2730
#define PROJ2    5460          // 2*PROJ
#define PROJ_PAD 2752          // PROJ rounded up to multiple of 64
#define EPSV     1e-5f

typedef __bf16 bf16;
typedef __bf16 v16bf __attribute__((ext_vector_type(16)));
typedef __bf16 v8bf  __attribute__((ext_vector_type(8)));
typedef float  v8f   __attribute__((ext_vector_type(8)));

__device__ __forceinline__ v16bf cat8(v8bf lo, v8bf hi) {
    return __builtin_shufflevector(lo, hi, 0,1,2,3,4,5,6,7,8,9,10,11,12,13,14,15);
}
__device__ __forceinline__ v8f wmma_bf16(v16bf a, v16bf b, v8f c) {
    return __builtin_amdgcn_wmma_f32_16x16x32_bf16(false, a, false, b, (short)0, c, false, false);
}
__device__ __forceinline__ bf16 f2bf(float f) { return (bf16)f; }

// CDNA5 async global->LDS copy, 16B per active lane, tracked by ASYNCcnt.
__device__ __forceinline__ void async_ld_b128(const void* gptr, void* lptr) {
    asm volatile("global_load_async_to_lds_b128 %0, %1, off"
                 :: "v"((unsigned)(size_t)lptr), "v"(gptr)
                 : "memory");
}
__device__ __forceinline__ void wait_async0() {
    asm volatile("s_wait_asynccnt 0" ::: "memory");
}

// A fragment: 16x32 bf16 (M x K). lanes 0-15: K[0:16); lanes 16-31: K[16:32)
__device__ __forceinline__ v16bf load_a_frag(const bf16* tile, int lane, int ldk) {
    int r = lane & 15, h = (lane >> 4) << 4;
    const bf16* p = tile + r * ldk + h;
    return cat8(*(const v8bf*)p, *(const v8bf*)(p + 8));
}
// B fragment from N-major LDS tile: lane -> column N (l&15), K half = (l>>4)*16
__device__ __forceinline__ v16bf load_b_frag(const bf16* tile, int lane, int ldk) {
    int r = lane & 15, h = (lane >> 4) << 4;
    const bf16* p = tile + r * ldk + h;
    return cat8(*(const v8bf*)p, *(const v8bf*)(p + 8));
}

// ---------------- weight transpose + f32->bf16 ----------------
// W: K x N f32 row-major  ->  Out: Npad x Kpad bf16, Out[n*Kpad+k] = (k<K&&n<N)? W[k*N+n] : 0
__global__ void __launch_bounds__(256)
transpose_convert_kernel(const float* __restrict__ W, bf16* __restrict__ Out,
                         int K, int N, int Kpad, int Npad) {
    __shared__ float tile[32][33];
    const int tx = threadIdx.x & 31, ty = threadIdx.x >> 5;
    const int k0 = blockIdx.x * 32, n0 = blockIdx.y * 32;
    #pragma unroll
    for (int i = ty; i < 32; i += 8) {
        int k = k0 + i, n = n0 + tx;
        tile[i][tx] = (k < K && n < N) ? W[(size_t)k * N + n] : 0.f;
    }
    __syncthreads();
    #pragma unroll
    for (int i = ty; i < 32; i += 8) {
        int n = n0 + i, k = k0 + tx;
        if (n < Npad && k < Kpad)
            Out[(size_t)n * Kpad + k] = f2bf(tile[tx][i]);
    }
}

// ---------------- layernorm(seq) -> bf16 X ----------------
__global__ void __launch_bounds__(256)
layernorm_kernel(const float* __restrict__ seq, const float* __restrict__ g,
                 const float* __restrict__ b, bf16* __restrict__ X) {
    __shared__ float s1[256], s2[256];
    const int t = threadIdx.x;
    const size_t base = (size_t)blockIdx.x * DIN;
    float v[8]; float s = 0.f, sq = 0.f;
    #pragma unroll
    for (int j = 0; j < 8; ++j) { v[j] = seq[base + t + j*256]; s += v[j]; sq += v[j]*v[j]; }
    s1[t] = s; s2[t] = sq;
    __syncthreads();
    for (int off = 128; off; off >>= 1) {
        if (t < off) { s1[t] += s1[t+off]; s2[t] += s2[t+off]; }
        __syncthreads();
    }
    const float mu  = s1[0] * (1.f / DIN);
    const float var = s2[0] * (1.f / DIN) - mu * mu;
    const float rs  = rsqrtf(var + EPSV);
    #pragma unroll
    for (int j = 0; j < 8; ++j) {
        int d = t + j*256;
        X[base + d] = f2bf((v[j] - mu) * rs * g[d] + b[d]);
    }
}

// ---------------- recurrent bias: rvec[g][e] = sum_d h0[h,d]*R_g[h,d,e] ----------------
__global__ void __launch_bounds__(256)
recur_kernel(const float* __restrict__ h0,
             const float* __restrict__ Ri, const float* __restrict__ Rf,
             const float* __restrict__ Rz, const float* __restrict__ Ro,
             float* __restrict__ rvec) {
    const int e = blockIdx.x * 256 + threadIdx.x;     // 0..D-1
    const int g = blockIdx.y;                         // gate: i,f,z,o
    const float* R = (g == 0) ? Ri : (g == 1) ? Rf : (g == 2) ? Rz : Ro;
    const int h = e >> 8, eh = e & 255;
    const float* hv = h0 + h * HDIM;
    const float* Rr = R + (size_t)h * HDIM * HDIM + eh;
    float acc = 0.f;
    for (int d = 0; d < HDIM; ++d) acc += hv[d] * Rr[(size_t)d * HDIM];
    rvec[g * DD + e] = acc;
}

// ---------------- fused 4-gate GEMM + sLSTM cell ----------------
// grid: (DD/64, BB/128), block 256 (8 waves, 4x2 wave grid, 32x32 per wave per gate)
__global__ void __launch_bounds__(256)
gates_kernel(const bf16* __restrict__ X,
             const bf16* __restrict__ Wti, const bf16* __restrict__ Wtf,
             const bf16* __restrict__ Wtz, const bf16* __restrict__ Wto,
             const float* __restrict__ bi, const float* __restrict__ bfv,
             const float* __restrict__ bz, const float* __restrict__ bo,
             const float* __restrict__ rvec,
             const float* __restrict__ c0, const float* __restrict__ n0,
             const float* __restrict__ m0,
             float* __restrict__ Cout, float* __restrict__ Nout,
             float* __restrict__ Hout, float* __restrict__ Mout) {
    __shared__ bf16 As[2][128][40];
    __shared__ bf16 Bs[2][4][64][40];
    const int t = threadIdx.x, lane = t & 31, wave = t >> 5;
    const int wm = wave >> 1, wn = wave & 1;
    const int bn = blockIdx.x * 64;
    const int bm = blockIdx.y * 128;
    const bf16* Wg[4] = {Wti, Wtf, Wtz, Wto};

    v8f acc[4][2][2];
    #pragma unroll
    for (int g = 0; g < 4; ++g)
        #pragma unroll
        for (int mi = 0; mi < 2; ++mi)
            #pragma unroll
            for (int ni = 0; ni < 2; ++ni) acc[g][mi][ni] = (v8f)0.0f;

    auto issue = [&](int p, int k0) {
        #pragma unroll
        for (int i = 0; i < 2; ++i) {               // A: 128x32 bf16
            int c = t + i * 256;
            int row = c >> 2, kc = (c & 3) << 3;
            async_ld_b128(X + (size_t)(bm + row) * DIN + k0 + kc, &As[p][row][kc]);
        }
        #pragma unroll
        for (int i = 0; i < 4; ++i) {               // 4 gates x 64x32 bf16 (N-major rows)
            int c = t + i * 256;
            int g = c >> 8, rem = c & 255;
            int nr = rem >> 2, kc = (rem & 3) << 3;
            async_ld_b128(Wg[g] + (size_t)(bn + nr) * DIN + k0 + kc, &Bs[p][g][nr][kc]);
        }
    };

    issue(0, 0);
    int p = 0;
    for (int k0 = 0; k0 < DIN; k0 += 32) {
        wait_async0();          // this wave's copies into buf[p] have landed
        __syncthreads();        // everyone's copies landed; prior readers done
        if (k0 + 32 < DIN) issue(p ^ 1, k0 + 32);   // prefetch under compute

        // preload ALL fragments first so ds_load latency is pipelined
        v16bf a[2];
        #pragma unroll
        for (int mi = 0; mi < 2; ++mi)
            a[mi] = load_a_frag(&As[p][wm*32 + mi*16][0], lane, 40);
        v16bf bfr[4][2];
        #pragma unroll
        for (int g = 0; g < 4; ++g)
            #pragma unroll
            for (int ni = 0; ni < 2; ++ni)
                bfr[g][ni] = load_b_frag(&Bs[p][g][wn*32 + ni*16][0], lane, 40);
        // 16 WMMAs back-to-back
        #pragma unroll
        for (int g = 0; g < 4; ++g)
            #pragma unroll
            for (int mi = 0; mi < 2; ++mi)
                #pragma unroll
                for (int ni = 0; ni < 2; ++ni)
                    acc[g][mi][ni] = wmma_bf16(a[mi], bfr[g][ni], acc[g][mi][ni]);
        __syncthreads();        // all reads of buf[p] done before it is refilled
        p ^= 1;
    }

    // epilogue: sLSTM cell math fused in registers
    const int nlo = lane & 15, mhi = (lane >> 4) << 3;
    #pragma unroll
    for (int ni = 0; ni < 2; ++ni) {
        const int n = bn + wn*32 + ni*16 + nlo;
        const float bi_n = bi[n]  + rvec[0*DD + n];
        const float bf_n = bfv[n] + rvec[1*DD + n];
        const float bz_n = bz[n]  + rvec[2*DD + n];
        const float bo_n = bo[n]  + rvec[3*DD + n];
        const float c0n = c0[n], n0n = n0[n], m0n = m0[n];
        #pragma unroll
        for (int mi = 0; mi < 2; ++mi) {
            #pragma unroll
            for (int r = 0; r < 8; ++r) {
                const int m = bm + wm*32 + mi*16 + mhi + r;
                float ip = acc[0][mi][ni][r] + bi_n;
                float fp = acc[1][mi][ni][r] + bf_n;
                float zp = acc[2][mi][ni][r] + bz_n;
                float op = acc[3][mi][ni][r] + bo_n;
                float mt = fmaxf(fp + m0n, ip);
                float it = __expf(ip - mt);
                float ft = __expf(fp - mt + m0n);
                float zt = tanhf(zp);
                float ot = 1.f / (1.f + __expf(-op));
                float ct = ft * c0n + it * zt;
                float nt = ft * n0n + it;
                float ht = ot * (ct / nt);
                size_t idx = (size_t)m * DD + n;
                Cout[idx] = ct; Nout[idx] = nt; Hout[idx] = ht; Mout[idx] = mt;
            }
        }
    }
}

// ---------------- groupnorm(h_t) -> bf16 XG ----------------
__global__ void __launch_bounds__(256)
groupnorm_kernel(const float* __restrict__ Ht, const float* __restrict__ g,
                 const float* __restrict__ b, bf16* __restrict__ XG) {
    const int lane = threadIdx.x & 31, w = threadIdx.x >> 5;
    const int hid = blockIdx.x * 8 + w;
    const int row = hid >> 3, head = hid & 7;
    const float* src = Ht + (size_t)row * DD + head * HDIM;
    float v[8]; float s = 0.f, sq = 0.f;
    #pragma unroll
    for (int j = 0; j < 8; ++j) { v[j] = src[lane + j*32]; s += v[j]; sq += v[j]*v[j]; }
    #pragma unroll
    for (int off = 16; off; off >>= 1) { s += __shfl_xor(s, off, 32); sq += __shfl_xor(sq, off, 32); }
    const float mu  = s  * (1.f / HDIM);
    const float var = sq * (1.f / HDIM) - mu * mu;
    const float rs  = rsqrtf(var + EPSV);
    bf16* dst = XG + (size_t)row * DD + head * HDIM;
    #pragma unroll
    for (int j = 0; j < 8; ++j) {
        int d = head * HDIM + lane + j*32;
        dst[lane + j*32] = f2bf((v[j] - mu) * rs * g[d] + b[d]);
    }
}

// ---------------- up GEMM: out1 + gelu(out2), two B tiles in one K loop ----------------
__global__ void __launch_bounds__(256)
up_kernel(const bf16* __restrict__ XG, const bf16* __restrict__ UT,
          const float* __restrict__ upb, bf16* __restrict__ G) {
    __shared__ bf16 As[2][128][40];
    __shared__ bf16 B1s[2][64][40];
    __shared__ bf16 B2s[2][64][40];
    const int t = threadIdx.x, lane = t & 31, wave = t >> 5;
    const int wm = wave >> 1, wn = wave & 1;
    const int bn = blockIdx.x * 64;
    const int bm = blockIdx.y * 128;

    // zero-fill B tiles once so rows beyond PROJ/PROJ2 contribute 0
    for (int i = t; i < 2 * 64 * 40; i += 256) {
        (&B1s[0][0][0])[i] = (bf16)0.f;
        (&B2s[0][0][0])[i] = (bf16)0.f;
    }
    __syncthreads();

    v8f a1[2][2], a2[2][2];
    #pragma unroll
    for (int mi = 0; mi < 2; ++mi)
        #pragma unroll
        for (int ni = 0; ni < 2; ++ni) { a1[mi][ni] = (v8f)0.0f; a2[mi][ni] = (v8f)0.0f; }

    auto issue = [&](int p, int k0) {
        #pragma unroll
        for (int i = 0; i < 2; ++i) {
            int c = t + i * 256;
            int row = c >> 2, kc = (c & 3) << 3;
            async_ld_b128(XG + (size_t)(bm + row) * DD + k0 + kc, &As[p][row][kc]);
        }
        {
            int nr = t >> 2, kc = (t & 3) << 3;
            int row1 = bn + nr;
            if (row1 < PROJ)
                async_ld_b128(UT + (size_t)row1 * DD + k0 + kc, &B1s[p][nr][kc]);
            int row2 = PROJ + bn + nr;
            if (row2 < PROJ2)
                async_ld_b128(UT + (size_t)row2 * DD + k0 + kc, &B2s[p][nr][kc]);
        }
    };

    issue(0, 0);
    int p = 0;
    for (int k0 = 0; k0 < DD; k0 += 32) {
        wait_async0();
        __syncthreads();
        if (k0 + 32 < DD) issue(p ^ 1, k0 + 32);

        v16bf a[2];
        #pragma unroll
        for (int mi = 0; mi < 2; ++mi)
            a[mi] = load_a_frag(&As[p][wm*32 + mi*16][0], lane, 40);
        v16bf b1[2], b2[2];
        #pragma unroll
        for (int ni = 0; ni < 2; ++ni) {
            b1[ni] = load_b_frag(&B1s[p][wn*32 + ni*16][0], lane, 40);
            b2[ni] = load_b_frag(&B2s[p][wn*32 + ni*16][0], lane, 40);
        }
        #pragma unroll
        for (int mi = 0; mi < 2; ++mi)
            #pragma unroll
            for (int ni = 0; ni < 2; ++ni) {
                a1[mi][ni] = wmma_bf16(a[mi], b1[ni], a1[mi][ni]);
                a2[mi][ni] = wmma_bf16(a[mi], b2[ni], a2[mi][ni]);
            }
        __syncthreads();
        p ^= 1;
    }

    const int nlo = lane & 15, mhi = (lane >> 4) << 3;
    #pragma unroll
    for (int ni = 0; ni < 2; ++ni) {
        const int n = bn + wn*32 + ni*16 + nlo;
        const bool valid = n < PROJ;
        const float bb1 = valid ? upb[n] : 0.f;
        const float bb2 = valid ? upb[PROJ + n] : 0.f;
        #pragma unroll
        for (int mi = 0; mi < 2; ++mi) {
            #pragma unroll
            for (int r = 0; r < 8; ++r) {
                const int m = bm + wm*32 + mi*16 + mhi + r;
                float x1 = a1[mi][ni][r] + bb1;
                float x2 = a2[mi][ni][r] + bb2;
                float gl = 0.5f * x2 * (1.f + erff(x2 * 0.70710678118654752f));
                G[(size_t)m * PROJ_PAD + n] = f2bf(x1 + gl);
            }
        }
    }
}

// ---------------- down GEMM + bias + residual ----------------
__global__ void __launch_bounds__(256)
down_kernel(const bf16* __restrict__ G, const bf16* __restrict__ DT,
            const float* __restrict__ db, const float* __restrict__ seq,
            float* __restrict__ Out) {
    __shared__ bf16 As[2][128][40];
    __shared__ bf16 Bs[2][64][40];
    const int t = threadIdx.x, lane = t & 31, wave = t >> 5;
    const int wm = wave >> 1, wn = wave & 1;
    const int bn = blockIdx.x * 64;
    const int bm = blockIdx.y * 128;

    v8f acc[2][2];
    #pragma unroll
    for (int mi = 0; mi < 2; ++mi)
        #pragma unroll
        for (int ni = 0; ni < 2; ++ni) acc[mi][ni] = (v8f)0.0f;

    auto issue = [&](int p, int k0) {
        #pragma unroll
        for (int i = 0; i < 2; ++i) {
            int c = t + i * 256;
            int row = c >> 2, kc = (c & 3) << 3;
            async_ld_b128(G + (size_t)(bm + row) * PROJ_PAD + k0 + kc, &As[p][row][kc]);
        }
        {
            int nr = t >> 2, kc = (t & 3) << 3;
            async_ld_b128(DT + (size_t)(bn + nr) * PROJ_PAD + k0 + kc, &Bs[p][nr][kc]);
        }
    };

    issue(0, 0);
    int p = 0;
    for (int k0 = 0; k0 < PROJ_PAD; k0 += 32) {
        wait_async0();
        __syncthreads();
        if (k0 + 32 < PROJ_PAD) issue(p ^ 1, k0 + 32);

        v16bf a[2], b[2];
        #pragma unroll
        for (int mi = 0; mi < 2; ++mi)
            a[mi] = load_a_frag(&As[p][wm*32 + mi*16][0], lane, 40);
        #pragma unroll
        for (int ni = 0; ni < 2; ++ni)
            b[ni] = load_b_frag(&Bs[p][wn*32 + ni*16][0], lane, 40);
        #pragma unroll
        for (int mi = 0; mi < 2; ++mi)
            #pragma unroll
            for (int ni = 0; ni < 2; ++ni)
                acc[mi][ni] = wmma_bf16(a[mi], b[ni], acc[mi][ni]);
        __syncthreads();
        p ^= 1;
    }

    const int nlo = lane & 15, mhi = (lane >> 4) << 3;
    #pragma unroll
    for (int ni = 0; ni < 2; ++ni) {
        const int n = bn + wn*32 + ni*16 + nlo;
        const float bn_v = db[n];
        #pragma unroll
        for (int mi = 0; mi < 2; ++mi) {
            #pragma unroll
            for (int r = 0; r < 8; ++r) {
                const int m = bm + wm*32 + mi*16 + mhi + r;
                size_t idx = (size_t)m * DIN + n;
                Out[idx] = acc[mi][ni][r] + bn_v + seq[idx];
            }
        }
    }
}

// ---------------- host launcher ----------------
extern "C" void kernel_launch(void* const* d_in, const int* in_sizes, int n_in,
                              void* d_out, int out_size, void* d_ws, size_t ws_size,
                              hipStream_t stream) {
    const float* seq   = (const float*)d_in[0];
    const float* c0    = (const float*)d_in[1];
    const float* n0    = (const float*)d_in[2];
    const float* h0    = (const float*)d_in[3];
    const float* m0    = (const float*)d_in[4];
    const float* ln_g  = (const float*)d_in[5];
    const float* ln_b  = (const float*)d_in[6];
    const float* gn_g  = (const float*)d_in[7];
    const float* gn_b  = (const float*)d_in[8];
    const float* Wz    = (const float*)d_in[9];
    const float* bz    = (const float*)d_in[10];
    const float* Wi    = (const float*)d_in[11];
    const float* bi    = (const float*)d_in[12];
    const float* Wo    = (const float*)d_in[13];
    const float* bo    = (const float*)d_in[14];
    const float* Wf    = (const float*)d_in[15];
    const float* bf_   = (const float*)d_in[16];
    const float* Rz    = (const float*)d_in[17];
    const float* Ri    = (const float*)d_in[18];
    const float* Ro    = (const float*)d_in[19];
    const float* Rf    = (const float*)d_in[20];
    const float* up_w  = (const float*)d_in[21];
    const float* up_b  = (const float*)d_in[22];
    const float* dn_w  = (const float*)d_in[23];
    const float* dn_b  = (const float*)d_in[24];

    float* Out  = (float*)d_out;
    float* Cout = Out  + (size_t)BB * DIN;
    float* Nout = Cout + (size_t)BB * DD;
    float* Hout = Nout + (size_t)BB * DD;
    float* Mout = Hout + (size_t)BB * DD;

    char* ws = (char*)d_ws;
    size_t off = 0;
    auto alloc = [&](size_t bytes) -> void* {
        void* p = ws + off;
        off += (bytes + 255) & ~(size_t)255;
        return p;
    };
    bf16* WTi   = (bf16*)alloc((size_t)DD * DIN * 2);
    bf16* WTf   = (bf16*)alloc((size_t)DD * DIN * 2);
    bf16* WTz   = (bf16*)alloc((size_t)DD * DIN * 2);
    bf16* WTo   = (bf16*)alloc((size_t)DD * DIN * 2);
    bf16* UT    = (bf16*)alloc((size_t)PROJ2 * DD * 2);
    bf16* DT    = (bf16*)alloc((size_t)DIN * PROJ_PAD * 2);
    bf16* Xbf   = (bf16*)alloc((size_t)BB * DIN * 2);
    bf16* XG    = (bf16*)alloc((size_t)BB * DD * 2);
    bf16* Gbuf  = (bf16*)alloc((size_t)BB * PROJ_PAD * 2);
    float* rvec = (float*)alloc((size_t)4 * DD * 4);

    const dim3 b256(256);

    // weight conversion + transpose to N-major bf16
    transpose_convert_kernel<<<dim3(DIN/32, DD/32), b256, 0, stream>>>(Wi, WTi, DIN, DD, DIN, DD);
    transpose_convert_kernel<<<dim3(DIN/32, DD/32), b256, 0, stream>>>(Wf, WTf, DIN, DD, DIN, DD);
    transpose_convert_kernel<<<dim3(DIN/32, DD/32), b256, 0, stream>>>(Wz, WTz, DIN, DD, DIN, DD);
    transpose_convert_kernel<<<dim3(DIN/32, DD/32), b256, 0, stream>>>(Wo, WTo, DIN, DD, DIN, DD);
    transpose_convert_kernel<<<dim3(DD/32, (PROJ2 + 31)/32), b256, 0, stream>>>(up_w, UT, DD, PROJ2, DD, PROJ2);
    transpose_convert_kernel<<<dim3(PROJ_PAD/32, DIN/32), b256, 0, stream>>>(dn_w, DT, PROJ, DIN, PROJ_PAD, DIN);

    // recurrent contribution (batch-invariant)
    recur_kernel<<<dim3(DD/256, 4), b256, 0, stream>>>(h0, Ri, Rf, Rz, Ro, rvec);

    // layernorm -> bf16
    layernorm_kernel<<<dim3(BB), b256, 0, stream>>>(seq, ln_g, ln_b, Xbf);

    // fused 4-gate GEMM + cell math -> c,n,h,m
    gates_kernel<<<dim3(DD/64, BB/128), b256, 0, stream>>>(
        Xbf, WTi, WTf, WTz, WTo, bi, bf_, bz, bo, rvec, c0, n0, m0,
        Cout, Nout, Hout, Mout);

    // groupnorm(h) -> bf16
    groupnorm_kernel<<<dim3(BB * HH / 8), b256, 0, stream>>>(Hout, gn_g, gn_b, XG);

    // up projection + gelu fuse
    up_kernel<<<dim3(PROJ_PAD/64, BB/128), b256, 0, stream>>>(XG, UT, up_b, Gbuf);

    // down projection + bias + residual
    down_kernel<<<dim3(DIN/64, BB/128), b256, 0, stream>>>(Gbuf, DT, dn_b, seq, Out);
}
